// Conv2DRand_53120155517322
// MI455X (gfx1250) — matrix operand order repaired and checked by reading, not compiled, not run on hardware
//
#include <hip/hip_runtime.h>

typedef __bf16 bf16;
typedef __attribute__((ext_vector_type(16))) __bf16 v16bf;
typedef __attribute__((ext_vector_type(8)))  __bf16 v8bf;
typedef __attribute__((ext_vector_type(8)))  float  v8f;

#define CH     64
#define HH     112
#define WW     112
#define NN     64
#define NPIX   (NN*HH*WW)        // 802816 pixels
#define S1B    256               // stage-1 reduction blocks
#define TCH    72                // padded channel stride in LDS tile (144B: 16B-aligned, conflict-free)
#define ROWS   4                 // output rows per workgroup (6 input rows in LDS, 1.5x read amp)
#define IROWS  (ROWS + 2)
#define TILE_ELEMS (IROWS * 114 * TCH)
#define CONV_LDS_BYTES (TILE_ELEMS * 2 + 512)

// round-to-nearest-even float -> bf16 (bit twiddle; no dependence on __bf16 conversion support)
__device__ __forceinline__ bf16 f2bf(float f) {
    union { float f; unsigned u; } in; in.f = f;
    unsigned u = in.u;
    unsigned r = (u + 0x7FFFu + ((u >> 16) & 1u)) >> 16;
    union { unsigned short s; bf16 b; } out; out.s = (unsigned short)r;
    return out.b;
}

union Frag { v16bf v; v8bf h[2]; };

// ---------------- Stage 1: per-channel partial sum / sumsq (deterministic, no atomics) ----------
__global__ __launch_bounds__(256) void stats_stage1(const float4* __restrict__ x4,
                                                    float* __restrict__ partial) {
    int tid = threadIdx.x;
    int c4  = tid & 15;     // channel quad 0..15  (channels 4*c4 .. 4*c4+3)
    int rw  = tid >> 4;     // pixel-row slot 0..15
    float4 s = make_float4(0.f, 0.f, 0.f, 0.f);
    float4 q = make_float4(0.f, 0.f, 0.f, 0.f);
    for (int row = blockIdx.x * 16 + rw; row < NPIX; row += gridDim.x * 16) {
        float4 v = x4[(long long)row * 16 + c4];
        s.x += v.x; s.y += v.y; s.z += v.z; s.w += v.w;
        q.x += v.x * v.x; q.y += v.y * v.y; q.z += v.z * v.z; q.w += v.w * v.w;
    }
    __shared__ float red[256][8];
    red[tid][0] = s.x; red[tid][1] = s.y; red[tid][2] = s.z; red[tid][3] = s.w;
    red[tid][4] = q.x; red[tid][5] = q.y; red[tid][6] = q.z; red[tid][7] = q.w;
    __syncthreads();
    if (rw == 0) {
        float acc[8] = {0.f, 0.f, 0.f, 0.f, 0.f, 0.f, 0.f, 0.f};
        for (int j = 0; j < 16; ++j) {
            const float* pr = red[c4 + 16 * j];
            for (int k = 0; k < 8; ++k) acc[k] += pr[k];
        }
        float* p = partial + (long long)blockIdx.x * 128;
        for (int u = 0; u < 4; ++u) {
            p[c4 * 4 + u]      = acc[u];      // sum
            p[64 + c4 * 4 + u] = acc[4 + u];  // sumsq
        }
    }
}

// ---------------- Stage 2: finalize fused BN scale/bias ----------------------------------------
__global__ __launch_bounds__(128) void stats_stage2(const float* __restrict__ partial,
                                                    const float* __restrict__ beta,
                                                    float* __restrict__ sb) {
    int t = threadIdx.x;                 // 0..127
    float acc = 0.f;
    for (int b = 0; b < S1B; ++b) acc += partial[(long long)b * 128 + t];
    __shared__ float s[128];
    s[t] = acc;
    __syncthreads();
    if (t < 64) {
        const float invM = 1.0f / (float)NPIX;
        float mean = s[t] * invM;
        float var  = s[64 + t] * invM - mean * mean;
        float sc   = rsqrtf(var + 1e-5f);
        sb[t]      = sc;                     // scale
        sb[64 + t] = beta[t] - mean * sc;    // bias
    }
}

// ---------------- Weight swizzle: HWIO fp32 -> per-lane-contiguous bf16 B-fragments ------------
// Fragment f = ((tap*2 + kk)*4 + nt). B 32x16 bf16 layout: lanes 0-15 -> N=lane, K{0..7,16..23};
// lanes 16-31 -> N=lane-16, K{8..15,24..31}. Each lane owns 16 contiguous bf16 (32 bytes).
__global__ __launch_bounds__(256) void wswizzle(const float* __restrict__ kern,
                                                bf16* __restrict__ wfrag) {
    int gid = blockIdx.x * blockDim.x + threadIdx.x;   // 0 .. 72*32-1
    if (gid >= 72 * 32) return;
    int lane = gid & 31;
    int f    = gid >> 5;
    int nt   = f & 3;
    int kk   = (f >> 2) & 1;
    int tap  = f >> 3;                                  // dy*3 + dx
    int co    = nt * 16 + (lane & 15);
    int kbase = (lane >= 16) ? 8 : 0;
    bf16* dst = wfrag + (long long)f * 512 + lane * 16;
    for (int e = 0; e < 16; ++e) {
        int ci = kk * 32 + kbase + e + ((e >= 8) ? 8 : 0);
        dst[e] = f2bf(kern[((long long)tap * 64 + ci) * 64 + co]);
    }
}

// ---------------- Fused BN + 3x3 SAME conv, WMMA bf16 implicit GEMM -----------------------------
// One workgroup (7 waves, 224 threads) = ROWS output rows at (n, h0..h0+ROWS-1):
// IROWS input rows staged once in LDS (bf16, BN fused), reused across the ROWS output rows.
__global__ __launch_bounds__(224) void conv_wmma(const float* __restrict__ x,
                                                 const bf16* __restrict__ wfrag,
                                                 const float* __restrict__ sb,
                                                 float* __restrict__ out) {
    extern __shared__ __attribute__((aligned(16))) char smem[];
    bf16*  tile = (bf16*)smem;                          // [IROWS][114][TCH]
    float* ssc  = (float*)(smem + TILE_ELEMS * 2);      // [64]
    float* sbi  = ssc + 64;                             // [64]

    const int tid = threadIdx.x;
    const int n   = blockIdx.x / (HH / ROWS);
    const int h0  = (blockIdx.x % (HH / ROWS)) * ROWS;

    if (tid < 128) {
        if (tid < 64) ssc[tid] = sb[tid];
        else          sbi[tid - 64] = sb[tid];
    }
    __syncthreads();

    // Cooperative load: input rows h0-1 .. h0+ROWS, cols -1..112 -> tile cols 0..113,
    // BN applied on the fly, halo x zero-filled before BN, so tile halo holds the bias.
    const float4* x4 = (const float4*)x;
    for (int i = tid; i < IROWS * 114 * 16; i += 224) {
        int r   = i / (114 * 16);
        int rem = i - r * (114 * 16);
        int col = rem >> 4;
        int cg  = rem & 15;
        int ih  = h0 + r - 1;
        int ic  = col - 1;
        float4 v = make_float4(0.f, 0.f, 0.f, 0.f);
        if ((unsigned)ih < (unsigned)HH && (unsigned)ic < (unsigned)WW) {
            v = x4[((long long)(n * HH + ih) * WW + ic) * 16 + cg];
        }
        int ch = cg * 4;
        bf16* t = tile + ((long long)r * 114 + col) * TCH + ch;
        t[0] = f2bf(v.x * ssc[ch + 0] + sbi[ch + 0]);
        t[1] = f2bf(v.y * ssc[ch + 1] + sbi[ch + 1]);
        t[2] = f2bf(v.z * ssc[ch + 2] + sbi[ch + 2]);
        t[3] = f2bf(v.w * ssc[ch + 3] + sbi[ch + 3]);
    }
    __syncthreads();

    const int wave = tid >> 5;         // 0..6, owns pixels [16*wave, 16*wave+16)
    const int lane = tid & 31;
    const int m    = lane & 15;
    const int hi   = lane >> 4;
    const int colbase = wave * 16 + m;

    const v8f vzero = {0.f, 0.f, 0.f, 0.f, 0.f, 0.f, 0.f, 0.f};

    for (int ro = 0; ro < ROWS; ++ro) {   // not unrolled: 72 static WMMA, 288 dynamic per wave
        v8f acc[4];
        acc[0] = vzero; acc[1] = vzero; acc[2] = vzero; acc[3] = vzero;

#pragma unroll
        for (int dy = 0; dy < 3; ++dy) {
#pragma unroll
            for (int dx = 0; dx < 3; ++dx) {
                const int tc = colbase + dx;     // output pixel p + dx maps to tile col p+dx
#pragma unroll
                for (int kk = 0; kk < 2; ++kk) {
                    // A 16x32 bf16 fragment: lanes<16 take K{0..7,16..23}, lanes>=16 K{8..15,24..31}
                    const bf16* ap = tile + ((long long)(ro + dy) * 114 + tc) * TCH
                                          + kk * 32 + hi * 8;
                    Frag a;
                    a.h[0] = *(const v8bf*)ap;
                    a.h[1] = *(const v8bf*)(ap + 16);
#pragma unroll
                    for (int nt = 0; nt < 4; ++nt) {
                        const int f = (((dy * 3 + dx) * 2 + kk) * 4) + nt;
                        const v8bf* bp = (const v8bf*)(wfrag + (long long)f * 512 + lane * 16);
                        Frag b;
                        b.h[0] = bp[0];
                        b.h[1] = bp[1];
                        acc[nt] = __builtin_amdgcn_wmma_f32_16x16x32_bf16(
                            false, a.v, false, b.v, (short)0, acc[nt], false, false);
                    }
                }
            }
        }

        // Epilogue: C/D 16x16 layout -> NHWC fp32. lane(N)=lane&15 (+16*nt), VGPR r -> M=r+8*hi.
        const long long rowbase = ((long long)(n * HH + (h0 + ro)) * WW) * CH;
#pragma unroll
        for (int nt = 0; nt < 4; ++nt) {
            const int nch = nt * 16 + (lane & 15);
#pragma unroll
            for (int r = 0; r < 8; ++r) {
                const int p = wave * 16 + r + hi * 8;
                out[rowbase + (long long)p * CH + nch] = acc[nt][r];
            }
        }
    }
}

// ---------------- host-side launcher ------------------------------------------------------------
extern "C" void kernel_launch(void* const* d_in, const int* in_sizes, int n_in,
                              void* d_out, int out_size, void* d_ws, size_t ws_size,
                              hipStream_t stream) {
    const float* x    = (const float*)d_in[0];   // (64,112,112,64) fp32 NHWC
    const float* kern = (const float*)d_in[1];   // (3,3,64,64) fp32 HWIO (+/-1)
    const float* beta = (const float*)d_in[2];   // (64,) fp32
    float* out = (float*)d_out;                  // (64,112,112,64) fp32

    // workspace layout: [0,73728) bf16 swizzled weight fragments
    //                   [73728,204800) stage-1 partials (256 blocks * 128 floats)
    //                   [204800,205312) fused scale/bias (128 floats)
    char*  ws      = (char*)d_ws;
    bf16*  wfrag   = (bf16*)ws;
    float* partial = (float*)(ws + 73728);
    float* sbuf    = (float*)(ws + 73728 + 131072);

    // conv uses ~99KB dynamic LDS (gfx1250: 320KB per WGP) -> opt in above the 64KB default.
    (void)hipFuncSetAttribute((const void*)conv_wmma,
                              hipFuncAttributeMaxDynamicSharedMemorySize,
                              CONV_LDS_BYTES);

    wswizzle<<<9, 256, 0, stream>>>(kern, wfrag);
    stats_stage1<<<S1B, 256, 0, stream>>>((const float4*)x, partial);
    stats_stage2<<<1, 128, 0, stream>>>(partial, beta, sbuf);
    conv_wmma<<<NN * (HH / ROWS), 224, 0, stream>>>(x, wfrag, sbuf, out);
}
